// LSTMDecoder_13761075216828
// MI455X (gfx1250) — matrix operand order repaired and checked by reading, not compiled, not run on hardware
//
#include <hip/hip_runtime.h>
#include <hip/hip_bf16.h>

constexpr int kB = 64;
constexpr int kS = 512;
constexpr int kH = 1024;
constexpr int kE = 512;
constexpr int kV = 32000;

typedef __attribute__((ext_vector_type(16))) __bf16 v16bf;
typedef __attribute__((ext_vector_type(8)))  float  v8f;

// Native conversion: backend emits v_cvt_pk_bf16_f32 (2 elems/op, RNE).
__device__ __forceinline__ __bf16 f2bf(float x) { return (__bf16)x; }

// Branch-free transcendentals on the VALU trans pipe (v_exp_f32 / v_rcp_f32)
// instead of the branchy libm expansions.
__device__ __forceinline__ float fast_rcp(float x) { return __builtin_amdgcn_rcpf(x); }
__device__ __forceinline__ float fast_sigmoid(float x) {
  return fast_rcp(1.f + __expf(-x));
}
__device__ __forceinline__ float fast_tanh(float x) {
  // 1 - 2/(e^{2x}+1); exact saturation at +-1 via inf/0 behavior of exp/rcp.
  return 1.f - 2.f * fast_rcp(__expf(2.f * x) + 1.f);
}

// ---------------------------------------------------------------------------
// C = A @ W^T (+bias / +=) via v_wmma_f32_16x16x32_bf16.
// A: (M,K) fp32 row-major. W: (N,K) fp32 row-major (PyTorch Linear layout).
// One wave computes a 16x64 tile: one A-frag feeds 4 WMMAs per K-step.
// grid = (N/64, M/16), block = 32. ACC/BIAS compile-time -> branch-free
// straight-line epilogue.
// ---------------------------------------------------------------------------
template <bool ACC, bool BIAS>
__global__ void wmma_gemm_atb(const float* __restrict__ A, const float* __restrict__ W,
                              const float* __restrict__ bias, float* __restrict__ C,
                              int Ndim, int Kdim) {
  const int lane  = threadIdx.x & 31;
  const int n0    = blockIdx.x * 64;
  const int m0    = blockIdx.y * 16;
  const int m     = lane & 15;
  const int hs    = lane >> 4;        // half-wave select
  const int col   = lane & 15;
  const int rbase = hs * 8;

  v8f c[4];
#pragma unroll
  for (int t = 0; t < 4; ++t) {
    if (ACC) {
#pragma unroll
      for (int r = 0; r < 8; ++r)
        c[t][r] = C[(size_t)(m0 + rbase + r) * Ndim + n0 + t * 16 + col];
    } else {
#pragma unroll
      for (int r = 0; r < 8; ++r) c[t][r] = 0.f;
    }
  }

  for (int k0 = 0; k0 < Kdim; k0 += 32) {
    // A-frag: lane = row m; halves 0..7 -> K=k0+hs*8+j, 8..15 -> K=k0+16+hs*8+j
    const float* ar = A + (size_t)(m0 + m) * Kdim + k0 + hs * 8;
    v16bf a;
#pragma unroll
    for (int j = 0; j < 8; ++j) { a[j] = f2bf(ar[j]); a[8 + j] = f2bf(ar[16 + j]); }

#pragma unroll
    for (int t = 0; t < 4; ++t) {
      // B-frag: lane = column n (row of W); halves 0..7 -> K=k0+hs*8+j, 8..15 -> K=k0+16+hs*8+j
      const float* wr = W + (size_t)(n0 + t * 16 + col) * Kdim + k0 + hs * 8;
      __builtin_prefetch(wr + 32, 0, 0);
      v16bf b;
#pragma unroll
      for (int j = 0; j < 8; ++j) { b[j] = f2bf(wr[j]); b[8 + j] = f2bf(wr[16 + j]); }
      c[t] = __builtin_amdgcn_wmma_f32_16x16x32_bf16(false, a, false, b, (short)0,
                                                     c[t], false, false);
    }
  }

#pragma unroll
  for (int t = 0; t < 4; ++t) {
#pragma unroll
    for (int r = 0; r < 8; ++r) {
      float out = c[t][r];
      if (BIAS) out += bias[n0 + t * 16 + col];
      C[(size_t)(m0 + rbase + r) * Ndim + n0 + t * 16 + col] = out;
    }
  }
}

// ---------------------------------------------------------------------------
// Fused attention energy: energy[b,s] = v . tanh(dproj[b] + enc[s,b] @ W2^T)
// One wave per 16-row s-tile. The 16x1024 A-tile is staged in LDS in WMMA
// *fragment order* (lane (m,hs) chunk = 16 contiguous halves), so each A-frag
// is one aligned 32B LDS read. 4 N-subtiles per pass share each A-frag.
// Never materializes the (B*S,H) intermediate (saves ~270 MB of traffic).
// grid = B*S/16, block = 32.
// ---------------------------------------------------------------------------
__global__ void attn_energy_kernel(const float* __restrict__ enc,    // (S,B,H)
                                   const float* __restrict__ W2,     // (H,H)
                                   const float* __restrict__ dproj,  // (B,H)
                                   const float* __restrict__ vvec,   // (H)
                                   float* __restrict__ energy) {     // (B,S)
  __shared__ __bf16 As[16 * kH];      // fragment-ordered, 32 KB
  const int lane = threadIdx.x;
  const int b  = blockIdx.x >> 5;     // S/16 == 32 tiles per batch row
  const int s0 = (blockIdx.x & 31) * 16;

  // Stage in fragment order: addr = ((blk*16 + m)*2 + hs)*16 + slot
  for (int i = lane; i < 16 * kH; i += 32) {
    int r = i >> 10;                  // row within tile
    int k = i & (kH - 1);
    int blk  = k >> 5;                // 32-wide K block
    int kk   = k & 31;
    int hsub = kk >> 4;
    int slot = kk & 15;
    As[((blk * 16 + r) * 2 + hsub) * 16 + slot] =
        f2bf(enc[(size_t)(s0 + r) * kB * kH + (size_t)b * kH + k]);
  }
  __syncthreads();

  const int m = lane & 15, hs = lane >> 4, col = lane & 15;
  float e[8];
#pragma unroll
  for (int r = 0; r < 8; ++r) e[r] = 0.f;

  for (int n0 = 0; n0 < kH; n0 += 64) {
    v8f c[4];
#pragma unroll
    for (int t = 0; t < 4; ++t)
#pragma unroll
      for (int r = 0; r < 8; ++r) c[t][r] = 0.f;

    for (int k0 = 0; k0 < kH; k0 += 32) {
      const int blk = k0 >> 5;
      v16bf a = *(const v16bf*)&As[((blk * 16 + m) * 2 + hs) * 16];

#pragma unroll
      for (int t = 0; t < 4; ++t) {
        const float* wr = W2 + (size_t)(n0 + t * 16 + col) * kH + k0 + hs * 16;
        __builtin_prefetch(wr + 32, 0, 0);
        v16bf bf;
#pragma unroll
        for (int j = 0; j < 16; ++j) bf[j] = f2bf(wr[j]);
        c[t] = __builtin_amdgcn_wmma_f32_16x16x32_bf16(false, a, false, bf, (short)0,
                                                       c[t], false, false);
      }
    }

#pragma unroll
    for (int t = 0; t < 4; ++t) {
      float dp = dproj[(size_t)b * kH + n0 + t * 16 + col];
      float vw = vvec[n0 + t * 16 + col];
#pragma unroll
      for (int r = 0; r < 8; ++r) e[r] += fast_tanh(c[t][r] + dp) * vw;
    }
  }

  // reduce over the 16 lanes of each half-wave (all hold the same rows)
#pragma unroll
  for (int off = 1; off < 16; off <<= 1) {
#pragma unroll
    for (int r = 0; r < 8; ++r) e[r] += __shfl_xor(e[r], off, 32);
  }
  if ((lane & 15) == 0) {
#pragma unroll
    for (int r = 0; r < 8; ++r)
      energy[(size_t)b * kS + s0 + hs * 8 + r] = e[r];
  }
}

// ---------------------------------------------------------------------------
__global__ void softmax_kernel(float* __restrict__ attn) {  // (B,S) in place
  __shared__ float red[256];
  const int b = blockIdx.x, tid = threadIdx.x;
  float* row = attn + (size_t)b * kS;
  float mx = -1e30f;
  for (int s = tid; s < kS; s += 256) mx = fmaxf(mx, row[s]);
  red[tid] = mx; __syncthreads();
  for (int off = 128; off > 0; off >>= 1) {
    if (tid < off) red[tid] = fmaxf(red[tid], red[tid + off]);
    __syncthreads();
  }
  mx = red[0]; __syncthreads();
  float sum = 0.f;
  for (int s = tid; s < kS; s += 256) { float t = __expf(row[s] - mx); row[s] = t; sum += t; }
  red[tid] = sum; __syncthreads();
  for (int off = 128; off > 0; off >>= 1) {
    if (tid < off) red[tid] += red[tid + off];
    __syncthreads();
  }
  float inv = fast_rcp(red[0]);
  for (int s = tid; s < kS; s += 256) row[s] *= inv;
}

// context[b,:] = sum_s attn[b,s] * enc[s,b,:]  (bandwidth bound: 134 MB pass)
// One block per batch row; attention row staged in LDS; float4 streaming.
__global__ void context_kernel(const float* __restrict__ attn, const float* __restrict__ enc,
                               float* __restrict__ ctx) {
  __shared__ float aw[kS];
  const int b = blockIdx.x;
  const int tid = threadIdx.x;
  for (int s = tid; s < kS; s += 256) aw[s] = attn[(size_t)b * kS + s];
  __syncthreads();
  const int h = tid * 4;                         // H/4 == 256 threads
  float4 acc = {0.f, 0.f, 0.f, 0.f};
  for (int s = 0; s < kS; ++s) {
    const float4 ev = *(const float4*)&enc[(size_t)s * kB * kH + (size_t)b * kH + h];
    const float w = aw[s];
    acc.x += w * ev.x; acc.y += w * ev.y; acc.z += w * ev.z; acc.w += w * ev.w;
  }
  *(float4*)&ctx[(size_t)b * kH + h] = acc;
}

__global__ void concat_x0_kernel(const int* __restrict__ tok, const float* __restrict__ emb,
                                 const float* __restrict__ ctx, float* __restrict__ x0) {
  int i = blockIdx.x * 256 + threadIdx.x;        // B*(E+H)
  int b = i / (kE + kH), j = i % (kE + kH);
  x0[i] = (j < kE) ? emb[(size_t)tok[b] * kE + j] : ctx[(size_t)b * kH + (j - kE)];
}

__global__ void lstm_pointwise_kernel(const float* __restrict__ gates,
                                      const float* __restrict__ bih, const float* __restrict__ bhh,
                                      const float* __restrict__ cprev,
                                      float* __restrict__ hout, float* __restrict__ cout) {
  int idx = blockIdx.x * 256 + threadIdx.x;      // B*H
  int b = idx >> 10, h = idx & (kH - 1);
  const float* g = gates + (size_t)b * 4 * kH;
  float gi = g[h]          + bih[h]          + bhh[h];
  float gf = g[kH + h]     + bih[kH + h]     + bhh[kH + h];
  float gg = g[2 * kH + h] + bih[2 * kH + h] + bhh[2 * kH + h];
  float go = g[3 * kH + h] + bih[3 * kH + h] + bhh[3 * kH + h];
  float i_ = fast_sigmoid(gi);
  float f_ = fast_sigmoid(gf);
  float o_ = fast_sigmoid(go);
  float g_ = fast_tanh(gg);
  float c2 = f_ * cprev[idx] + i_ * g_;
  hout[idx] = o_ * fast_tanh(c2);
  cout[idx] = c2;
}

__global__ void concat_y_kernel(const float* __restrict__ h1, const float* __restrict__ ctx,
                                float* __restrict__ y) {
  int i = blockIdx.x * 256 + threadIdx.x;        // B*2H
  int b = i / (2 * kH), j = i % (2 * kH);
  y[i] = (j < kH) ? h1[(size_t)b * kH + j] : ctx[(size_t)b * kH + (j - kH)];
}

// ---------------------------------------------------------------------------
extern "C" void kernel_launch(void* const* d_in, const int* in_sizes, int n_in,
                              void* d_out, int out_size, void* d_ws, size_t ws_size,
                              hipStream_t stream) {
  const int*   tok  = (const int*)  d_in[0];
  const float* hid  = (const float*)d_in[1];   // (L,B,H)
  const float* cel  = (const float*)d_in[2];   // (L,B,H)
  const float* enc  = (const float*)d_in[3];   // (S,B,H)
  const float* emb  = (const float*)d_in[4];   // (V,E)
  const float* W1   = (const float*)d_in[5];
  const float* W2   = (const float*)d_in[6];
  const float* vv   = (const float*)d_in[7];
  const float* Wih0 = (const float*)d_in[8];
  const float* Whh0 = (const float*)d_in[9];
  const float* bih0 = (const float*)d_in[10];
  const float* bhh0 = (const float*)d_in[11];
  const float* Wih1 = (const float*)d_in[12];
  const float* Whh1 = (const float*)d_in[13];
  const float* bih1 = (const float*)d_in[14];
  const float* bhh1 = (const float*)d_in[15];
  const float* fcw  = (const float*)d_in[16];
  const float* fcb  = (const float*)d_in[17];

  float* ws     = (float*)d_ws;
  float* dproj  = ws;                       // B*H
  float* attn   = dproj  + kB * kH;         // B*S
  float* ctx    = attn   + kB * kS;         // B*H
  float* x0     = ctx    + kB * kH;         // B*(E+H)
  float* gates0 = x0     + kB * (kE + kH);  // B*4H
  float* gates1 = gates0 + kB * 4 * kH;     // B*4H
  float* ycat   = gates1 + kB * 4 * kH;     // B*2H

  float* logits = (float*)d_out;                 // B*V
  float* h0o = logits + (size_t)kB * kV;         // hidden_new[0]
  float* h1o = h0o + kB * kH;                    // hidden_new[1]
  float* c0o = h1o + kB * kH;                    // cell_new[0]
  float* c1o = c0o + kB * kH;                    // cell_new[1]

  // 1. dproj = hidden[-1] @ W1^T                       (64 x 1024 x 1024)
  wmma_gemm_atb<false, false><<<dim3(kH / 64, kB / 16), 32, 0, stream>>>(
      hid + kB * kH, W1, nullptr, dproj, kH, kH);
  // 2. fused energy = v . tanh(dproj + enc @ W2^T)     (the 68.7 GF hot loop)
  attn_energy_kernel<<<kB * (kS / 16), 32, 0, stream>>>(enc, W2, dproj, vv, attn);
  // 3. softmax over S
  softmax_kernel<<<kB, 256, 0, stream>>>(attn);
  // 4. context = attn . enc                            (134 MB streaming pass)
  context_kernel<<<kB, 256, 0, stream>>>(attn, enc, ctx);
  // 5. x0 = [embedding[tok], context]
  concat_x0_kernel<<<kB * (kE + kH) / 256, 256, 0, stream>>>(tok, emb, ctx, x0);
  // 6-7. gates0 = x0 @ Wih0^T + h[0] @ Whh0^T
  wmma_gemm_atb<false, false><<<dim3(4 * kH / 64, kB / 16), 32, 0, stream>>>(
      x0, Wih0, nullptr, gates0, 4 * kH, kE + kH);
  wmma_gemm_atb<true, false><<<dim3(4 * kH / 64, kB / 16), 32, 0, stream>>>(
      hid, Whh0, nullptr, gates0, 4 * kH, kH);
  // 8. layer-0 pointwise
  lstm_pointwise_kernel<<<kB * kH / 256, 256, 0, stream>>>(gates0, bih0, bhh0, cel, h0o, c0o);
  // 9-10. gates1 = h0 @ Wih1^T + h[1] @ Whh1^T
  wmma_gemm_atb<false, false><<<dim3(4 * kH / 64, kB / 16), 32, 0, stream>>>(
      h0o, Wih1, nullptr, gates1, 4 * kH, kH);
  wmma_gemm_atb<true, false><<<dim3(4 * kH / 64, kB / 16), 32, 0, stream>>>(
      hid + kB * kH, Whh1, nullptr, gates1, 4 * kH, kH);
  // 11. layer-1 pointwise
  lstm_pointwise_kernel<<<kB * kH / 256, 256, 0, stream>>>(gates1, bih1, bhh1,
                                                           cel + kB * kH, h1o, c1o);
  // 12. y = [h1, context]
  concat_y_kernel<<<kB * 2 * kH / 256, 256, 0, stream>>>(h1o, ctx, ycat);
  // 13. logits = y @ fc_w^T + fc_b                      (64 x 32000 x 2048)
  wmma_gemm_atb<false, true><<<dim3(kV / 64, kB / 16), 32, 0, stream>>>(
      ycat, fcw, fcb, logits, kV, 2 * kH);
}
